// TBlock_45603962749200
// MI455X (gfx1250) — matrix operand order repaired and checked by reading, not compiled, not run on hardware
//
#include <hip/hip_runtime.h>
#include <hip/hip_bf16.h>
#include <math.h>

// Problem constants (match reference)
#define B_   4
#define S_   2048
#define H_   512
#define NH_  8
#define HD_  64
#define MLP_ 2048
#define ROWS (B_*S_)   // 8192

typedef __attribute__((ext_vector_type(16))) __bf16       v16bf;
typedef __attribute__((ext_vector_type(8)))  float        v8f;
typedef __attribute__((ext_vector_type(4)))  unsigned int u32x4;

union FragA { u32x4 q[2]; v16bf v; };

__device__ __forceinline__ unsigned short f2bf(float f) {
  unsigned u = __float_as_uint(f);
  u += 0x7FFFu + ((u >> 16) & 1u);       // round-to-nearest-even
  return (unsigned short)(u >> 16);
}

__device__ __forceinline__ v8f wmma_bf16(v16bf a, v16bf b, v8f c) {
  return __builtin_amdgcn_wmma_f32_16x16x32_bf16(false, a, false, b,
                                                 (short)0, c, false, false);
}

// ---------------------------------------------------------------------------
// Async global->LDS staging (gfx1250 GLOBAL_LOAD_ASYNC_TO_LDS_B128, ASYNCcnt).
// Builtin signature (from hipcc diagnostic): (int4 AS1*, int4 AS3*, Imm, Imm).
// ---------------------------------------------------------------------------
#if __has_builtin(__builtin_amdgcn_global_load_async_to_lds_b128) && \
    __has_builtin(__builtin_amdgcn_s_wait_asynccnt)
#define ASYNC_LDS 1
typedef __attribute__((ext_vector_type(4))) int v4i_;
typedef __attribute__((address_space(1))) v4i_ v4i_glob;
typedef __attribute__((address_space(3))) v4i_ v4i_lds;
#define WAIT_ASYNC(n) __builtin_amdgcn_s_wait_asynccnt(n)
#else
#define ASYNC_LDS 0
#define WAIT_ASYNC(n)
#endif

__device__ __forceinline__ void cp16(unsigned short* lds, const unsigned short* g) {
#if ASYNC_LDS
  __builtin_amdgcn_global_load_async_to_lds_b128(
      (v4i_glob*)(void*)g, (v4i_lds*)(void*)lds, 0, 0);
#else
  *(u32x4*)lds = *(const u32x4*)g;
#endif
}

// ---------------------------------------------------------------------------
// Tensor Data Mover: 2D tile (bf16) global -> LDS, ISA ch.8 descriptor layout.
// This toolchain's builtin arity (from hipcc diagnostic): 6 args
//   (uint32x4 g0, int32x8 g1, int32x4, int32x4, int32x8, i32 cpol).
// ---------------------------------------------------------------------------
#if ASYNC_LDS && __has_builtin(__builtin_amdgcn_tensor_load_to_lds) && \
    __has_builtin(__builtin_amdgcn_s_wait_tensorcnt)
#define HAS_TDM 1
typedef __attribute__((ext_vector_type(8))) int i32x8;
typedef __attribute__((ext_vector_type(4))) int i32x4;
#define WAIT_TENSOR(n) __builtin_amdgcn_s_wait_tensorcnt(n)

// tile_d0 elements per row (dim0, contiguous), tile_d1 rows; element = 2 bytes.
// Tile is fully in-bounds; rows packed consecutively in LDS (no padding).
__device__ __forceinline__ void tdm_load_2d(
    const unsigned short* gbase, unsigned lds_byte_off,
    unsigned tensor_d0, unsigned tensor_d1,
    unsigned tile_d0, unsigned tile_d1, unsigned row_stride_elems)
{
  unsigned long long ga = (unsigned long long)(size_t)gbase;
  u32x4 g0;
  g0[0] = 1u;                                               // count=1 (valid D#)
  g0[1] = lds_byte_off;                                     // lds_addr
  g0[2] = (unsigned)(ga & 0xFFFFFFFFu);                     // global_addr[31:0]
  g0[3] = (unsigned)((ga >> 32) & 0x01FFFFFFu) | (2u << 30);// addr[56:32] | type=2
  i32x8 g1;
  g1[0] = (int)(1u << 16);                                  // data_size=1 (2B)
  g1[1] = (int)((tensor_d0 & 0xFFFFu) << 16);               // tensor_dim0[15:0]
  g1[2] = (int)((tensor_d0 >> 16) | ((tensor_d1 & 0xFFFFu) << 16)); // d0 hi | d1 lo
  g1[3] = (int)((tensor_d1 >> 16) | (tile_d0 << 16));       // d1 hi | tile_dim0
  g1[4] = (int)(tile_d1 & 0xFFFFu);                         // tile_dim1 (tile_dim2=0)
  g1[5] = (int)row_stride_elems;                            // dim0_stride[31:0]
  g1[6] = 0;                                                // stride hi | d1_stride lo
  g1[7] = 0;
  i32x4 z4 = {0, 0, 0, 0};                                  // groups 2/3 unused (2D)
  i32x8 z8 = {0, 0, 0, 0, 0, 0, 0, 0};
  __builtin_amdgcn_tensor_load_to_lds(g0, g1, z4, z4, z8, 0);
}

__device__ __forceinline__ unsigned lds_offset_of(void* p) {
  return (unsigned)(unsigned long long)(v4i_lds*)p;
}
#else
#define HAS_TDM 0
#define WAIT_TENSOR(n)
#endif

// ---------------------------------------------------------------------------
// Weight convert+transpose: wt[(rowoff+n)*K + k] = bf16(w[k*N + n] * scale).
// K-major bf16 weights so GEMM B-tiles are contiguous-in-K (TDM/async b128
// friendly). Query scale folded into Wq.
// ---------------------------------------------------------------------------
__global__ __launch_bounds__(256) void cvt_wt(
    const float* __restrict__ w, unsigned short* __restrict__ wt,
    int K, int N, int rowoff, float scale)
{
  const size_t total = (size_t)K * N;
  for (size_t idx = (size_t)blockIdx.x * 256 + threadIdx.x; idx < total;
       idx += (size_t)gridDim.x * 256) {
    int k = (int)(idx / N), n = (int)(idx % N);
    wt[(size_t)(rowoff + n) * K + k] = f2bf(w[idx] * scale);
  }
}

// ---------------------------------------------------------------------------
// LayerNorm over H=512, one row per block (256 threads, 8 waves, 2 elem/thr).
// ---------------------------------------------------------------------------
__global__ __launch_bounds__(256) void ln_rows(
    const float* __restrict__ x, const float* __restrict__ mask,
    unsigned short* __restrict__ y, int applyMask)
{
  __shared__ float red[8];
  const int row = blockIdx.x, tid = threadIdx.x;
  const float mv = applyMask ? mask[row] : 1.0f;
  float v0 = x[(size_t)row*H_ + tid]       * mv;
  float v1 = x[(size_t)row*H_ + 256 + tid] * mv;

  float s = v0 + v1;
  #pragma unroll
  for (int o = 16; o; o >>= 1) s += __shfl_xor(s, o, 32);
  if ((tid & 31) == 0) red[tid >> 5] = s;
  __syncthreads();
  float tot = 0.f;
  #pragma unroll
  for (int i = 0; i < 8; i++) tot += red[i];
  const float mean = tot * (1.0f / H_);
  __syncthreads();

  const float d0 = v0 - mean, d1 = v1 - mean;
  s = d0*d0 + d1*d1;
  #pragma unroll
  for (int o = 16; o; o >>= 1) s += __shfl_xor(s, o, 32);
  if ((tid & 31) == 0) red[tid >> 5] = s;
  __syncthreads();
  tot = 0.f;
  #pragma unroll
  for (int i = 0; i < 8; i++) tot += red[i];
  const float r = rsqrtf(tot * (1.0f / H_) + 1e-6f);

  y[(size_t)row*H_ + tid]       = f2bf(d0 * r);
  y[(size_t)row*H_ + 256 + tid] = f2bf(d1 * r);
}

// ---------------------------------------------------------------------------
// Tiled WMMA GEMM:  C[M,N] = A[M,K](bf16) * W[K,N]  (W given K-major as Wt[N][K])
// Block tile 256x128, K-step 32, double-buffered. A-tiles staged with async
// b128 loads; B-tiles (weights) staged by the Tensor Data Mover (wave 0 issues
// one TENSOR_LOAD_TO_LDS per tile). 8 waves in 4(M)x2(N); each wave computes a
// 64x64 tile -> 16 v_wmma per K-step vs 8 fragment loads.
// Epilogues: 0 = scale -> bf16   1 = exact GELU -> bf16
//            2 = +resid, *mask[row] -> f32   3 = +resid -> f32
// ---------------------------------------------------------------------------
template<int EPI>
__global__ __launch_bounds__(256) void gemm_wmma(
    const unsigned short* __restrict__ A, int lda,
    const unsigned short* __restrict__ Wt,   // [N][K] bf16
    void* __restrict__ Cout, int ldc,
    int M, int N, int K,
    const float* __restrict__ resid,
    const float* __restrict__ mask,
    float scale)
{
  __shared__ __align__(16) unsigned short a_lds[2][256][32];  // [buf][m][k]
  __shared__ __align__(16) unsigned short b_lds[2][128][32];  // [buf][n][k]

  const int tid   = threadIdx.x;
  const int wave  = tid >> 5, lane = tid & 31;
  const int lmod  = lane & 15, lhalf = lane >> 4;
  const int bm    = blockIdx.y * 256, bn = blockIdx.x * 128;
  const int wm    = (wave >> 1) * 64, wn = (wave & 1) * 64;

  v8f acc[4][4] = {};

  auto stage = [&](int p, int k0) {
    #pragma unroll
    for (int i = 0; i < 4; i++) {            // A: 4 async b128 per thread
      int c = tid + i * 256, row = c >> 2, c4 = c & 3;
      cp16(&a_lds[p][row][c4 * 8], A + (size_t)(bm + row) * lda + k0 + c4 * 8);
    }
#if HAS_TDM
    if (wave == 0)                            // B: one TDM 2D tile per buffer
      tdm_load_2d(Wt + (size_t)bn * K + k0, lds_offset_of(&b_lds[p][0][0]),
                  (unsigned)K, 128u, 32u, 128u, (unsigned)K);
#else
    #pragma unroll
    for (int i = 0; i < 2; i++) {
      int c = tid + i * 256, row = c >> 2, c4 = c & 3;
      cp16(&b_lds[p][row][c4 * 8], Wt + (size_t)(bn + row) * K + k0 + c4 * 8);
    }
#endif
  };

  stage(0, 0);
  int pbuf = 0;
  for (int k0 = 0; k0 < K; k0 += 32) {
    if (k0 + 32 < K) {
      stage(pbuf ^ 1, k0 + 32);
#if HAS_TDM
      WAIT_ASYNC(4);
      if (wave == 0) WAIT_TENSOR(1);
#else
      WAIT_ASYNC(6);
#endif
    } else {
      WAIT_ASYNC(0);
      if (wave == 0) WAIT_TENSOR(0);
    }
    __syncthreads();                       // staging of pbuf complete (RAW)

    FragA af[4], bfr[4];
    #pragma unroll
    for (int mt = 0; mt < 4; mt++) {
      int m = wm + mt * 16 + lmod;
      af[mt].q[0] = *(const u32x4*)&a_lds[pbuf][m][lhalf * 8];
      af[mt].q[1] = *(const u32x4*)&a_lds[pbuf][m][16 + lhalf * 8];
    }
    #pragma unroll
    for (int nt = 0; nt < 4; nt++) {
      int n = wn + nt * 16 + lmod;
      bfr[nt].q[0] = *(const u32x4*)&b_lds[pbuf][n][lhalf * 8];
      bfr[nt].q[1] = *(const u32x4*)&b_lds[pbuf][n][16 + lhalf * 8];
    }
    #pragma unroll
    for (int mt = 0; mt < 4; mt++)
      #pragma unroll
      for (int nt = 0; nt < 4; nt++)
        acc[mt][nt] = wmma_bf16(af[mt].v, bfr[nt].v, acc[mt][nt]);

    __syncthreads();                       // all reads of pbuf done (WAR)
    pbuf ^= 1;
  }

  // Epilogue (C layout: lane n = lmod, VGPR j -> row j + 8*lhalf)
  #pragma unroll
  for (int mt = 0; mt < 4; mt++)
    #pragma unroll
    for (int nt = 0; nt < 4; nt++)
      #pragma unroll
      for (int j = 0; j < 8; j++) {
        int row = bm + wm + mt * 16 + lhalf * 8 + j;
        int col = bn + wn + nt * 16 + lmod;
        float x = acc[mt][nt][j];
        if (EPI == 0) {
          ((unsigned short*)Cout)[(size_t)row * ldc + col] = f2bf(x * scale);
        } else if (EPI == 1) {
          float g = 0.5f * x * (1.0f + erff(x * 0.70710678118654752f));
          ((unsigned short*)Cout)[(size_t)row * ldc + col] = f2bf(g);
        } else if (EPI == 2) {
          float o = (x + resid[(size_t)row * ldc + col]) * mask[row];
          ((float*)Cout)[(size_t)row * ldc + col] = o;
        } else {
          float o = x + resid[(size_t)row * ldc + col];
          ((float*)Cout)[(size_t)row * ldc + col] = o;
        }
      }
}

// ---------------------------------------------------------------------------
// Flash attention: one block = (64 query rows) x (one b,h). 128 threads,
// 4 waves; wave w owns query rows [w*16, w*16+16). Streams keys in 64-chunks.
// Q/K tiles staged via async-to-LDS. Query-side mask is a row-constant logit
// shift -> softmax invariant -> omitted (matches reference semantics).
// ---------------------------------------------------------------------------
__global__ __launch_bounds__(128) void attn_flash(
    const unsigned short* __restrict__ qkv,   // [ROWS][1536] bf16 (q|k|v)
    unsigned short* __restrict__ out)         // [ROWS][512] bf16
{
  __shared__ __align__(16) unsigned short q_lds[64][64];
  __shared__ __align__(16) unsigned short k_lds[64][64];   // [key][hd]
  __shared__ __align__(16) unsigned short vt_lds[64][64];  // [hd][key]
  __shared__ __align__(16) unsigned short p_lds[4][16][64];

  const int tid  = threadIdx.x;
  const int wave = tid >> 5, lane = tid & 31;
  const int lmod = lane & 15, lhalf = lane >> 4;
  const int q0 = blockIdx.x * 64;
  const int h  = blockIdx.y, b = blockIdx.z;

  // Stage Q tile once (async)
  #pragma unroll
  for (int i = 0; i < 4; i++) {
    int c = tid + i * 128; int row = c >> 3, c8 = c & 7;
    cp16(&q_lds[row][c8 * 8],
         qkv + (size_t)(b * S_ + q0 + row) * 1536 + h * 64 + c8 * 8);
  }

  float run_m[8], run_l[8];
  #pragma unroll
  for (int j = 0; j < 8; j++) { run_m[j] = -3e38f; run_l[j] = 0.f; }
  v8f o[4] = {};

  for (int kc0 = 0; kc0 < S_; kc0 += 64) {
    __syncthreads();  // protect k/vt from previous iteration's readers
    #pragma unroll
    for (int i = 0; i < 4; i++) {
      int c = tid + i * 128; int row = c >> 3, c8 = c & 7;
      cp16(&k_lds[row][c8 * 8],
           qkv + (size_t)(b * S_ + kc0 + row) * 1536 + 512 + h * 64 + c8 * 8);
    }
    #pragma unroll
    for (int i = 0; i < 32; i++) {
      int e = tid + i * 128; int key = e >> 6, hd = e & 63;
      vt_lds[hd][key] = qkv[(size_t)(b * S_ + kc0 + key) * 1536 + 1024 + h * 64 + hd];
    }
    WAIT_ASYNC(0);
    __syncthreads();

    // S = Q * K^T  (16 rows x 64 keys per wave), K-chain over HD=64
    v8f s[4] = {};
    #pragma unroll
    for (int kc = 0; kc < 2; kc++) {
      FragA aq;
      int m = wave * 16 + lmod;
      aq.q[0] = *(const u32x4*)&q_lds[m][kc * 32 + lhalf * 8];
      aq.q[1] = *(const u32x4*)&q_lds[m][kc * 32 + 16 + lhalf * 8];
      #pragma unroll
      for (int nt = 0; nt < 4; nt++) {
        FragA bk;
        int n = nt * 16 + lmod;
        bk.q[0] = *(const u32x4*)&k_lds[n][kc * 32 + lhalf * 8];
        bk.q[1] = *(const u32x4*)&k_lds[n][kc * 32 + 16 + lhalf * 8];
        s[nt] = wmma_bf16(aq.v, bk.v, s[nt]);
      }
    }

    // Streaming softmax per row (row j+8*lhalf lives across 16 lanes)
    #pragma unroll
    for (int j = 0; j < 8; j++) {
      float mj = s[0][j];
      mj = fmaxf(mj, s[1][j]); mj = fmaxf(mj, s[2][j]); mj = fmaxf(mj, s[3][j]);
      #pragma unroll
      for (int off = 8; off; off >>= 1) mj = fmaxf(mj, __shfl_xor(mj, off, 16));
      float nm    = fmaxf(run_m[j], mj);
      float alpha = __expf(run_m[j] - nm);
      run_m[j] = nm;
      float rs = 0.f;
      #pragma unroll
      for (int nt = 0; nt < 4; nt++) {
        float pv = __expf(s[nt][j] - nm);
        rs += pv;
        p_lds[wave][lhalf * 8 + j][nt * 16 + lmod] = f2bf(pv);
      }
      #pragma unroll
      for (int off = 8; off; off >>= 1) rs += __shfl_xor(rs, off, 16);
      run_l[j] = run_l[j] * alpha + rs;
      #pragma unroll
      for (int nt = 0; nt < 4; nt++) o[nt][j] *= alpha;
    }
    __syncthreads();  // P visible (and orders LDS traffic)

    // O += P * V
    #pragma unroll
    for (int kc = 0; kc < 2; kc++) {
      FragA pa;
      pa.q[0] = *(const u32x4*)&p_lds[wave][lmod][kc * 32 + lhalf * 8];
      pa.q[1] = *(const u32x4*)&p_lds[wave][lmod][kc * 32 + 16 + lhalf * 8];
      #pragma unroll
      for (int nt = 0; nt < 4; nt++) {
        FragA vb;
        int n = nt * 16 + lmod;
        vb.q[0] = *(const u32x4*)&vt_lds[n][kc * 32 + lhalf * 8];
        vb.q[1] = *(const u32x4*)&vt_lds[n][kc * 32 + 16 + lhalf * 8];
        o[nt] = wmma_bf16(pa.v, vb.v, o[nt]);
      }
    }
  }

  // Normalize and store (heads re-merged: col = h*64 + hd)
  #pragma unroll
  for (int nt = 0; nt < 4; nt++)
    #pragma unroll
    for (int j = 0; j < 8; j++) {
      int row = q0 + wave * 16 + lhalf * 8 + j;
      float val = o[nt][j] / run_l[j];
      out[(size_t)(b * S_ + row) * 512 + h * 64 + nt * 16 + lmod] = f2bf(val);
    }
}

// ---------------------------------------------------------------------------
// Launcher
// ---------------------------------------------------------------------------
extern "C" void kernel_launch(void* const* d_in, const int* in_sizes, int n_in,
                              void* d_out, int out_size, void* d_ws, size_t ws_size,
                              hipStream_t stream) {
  const float* input = (const float*)d_in[0];
  const float* mask  = (const float*)d_in[1];
  const float* Wq    = (const float*)d_in[2];
  const float* Wk    = (const float*)d_in[3];
  const float* Wv    = (const float*)d_in[4];
  const float* Wo    = (const float*)d_in[5];
  const float* W1    = (const float*)d_in[6];
  const float* W2    = (const float*)d_in[7];

  char* ws = (char*)d_ws;
  size_t off = 0;
  unsigned short* xln    = (unsigned short*)(ws + off); off += (size_t)ROWS*H_*2;     // 8 MiB
  unsigned short* qkv    = (unsigned short*)(ws + off); off += (size_t)ROWS*1536*2;   // 24 MiB
  unsigned short* att    = (unsigned short*)(ws + off); off += (size_t)ROWS*H_*2;     // 8 MiB
  float*          x2     = (float*)         (ws + off); off += (size_t)ROWS*H_*4;     // 16 MiB
  unsigned short* y      = (unsigned short*)(ws + off); off += (size_t)ROWS*H_*2;     // 8 MiB
  unsigned short* h1     = (unsigned short*)(ws + off); off += (size_t)ROWS*MLP_*2;   // 32 MiB
  unsigned short* wqkv_t = (unsigned short*)(ws + off); off += (size_t)1536*H_*2;     // 1.5 MiB
  unsigned short* wo_t   = (unsigned short*)(ws + off); off += (size_t)H_*H_*2;       // 0.5 MiB
  unsigned short* w1_t   = (unsigned short*)(ws + off); off += (size_t)MLP_*H_*2;     // 2 MiB
  unsigned short* w2_t   = (unsigned short*)(ws + off); off += (size_t)H_*MLP_*2;     // 2 MiB

  // 0) Convert weights to K-major bf16 (q-scale folded into Wq)
  cvt_wt<<<1024, 256, 0, stream>>>(Wq, wqkv_t, H_, H_,    0, 0.125f);
  cvt_wt<<<1024, 256, 0, stream>>>(Wk, wqkv_t, H_, H_,  512, 1.0f);
  cvt_wt<<<1024, 256, 0, stream>>>(Wv, wqkv_t, H_, H_, 1024, 1.0f);
  cvt_wt<<<1024, 256, 0, stream>>>(Wo, wo_t,   H_, H_,    0, 1.0f);
  cvt_wt<<<2048, 256, 0, stream>>>(W1, w1_t,   H_, MLP_,  0, 1.0f);
  cvt_wt<<<2048, 256, 0, stream>>>(W2, w2_t,   MLP_, H_,  0, 1.0f);

  // 1) x = LN(input * mask)  -> bf16
  ln_rows<<<ROWS, 256, 0, stream>>>(input, mask, xln, 1);

  // 2) fused q|k|v projection into [ROWS][1536]
  gemm_wmma<0><<<dim3(12, 32), 256, 0, stream>>>(xln, H_, wqkv_t, qkv, 1536,
      ROWS, 1536, H_, nullptr, nullptr, 1.0f);

  // 3) flash attention -> att [ROWS][512] bf16
  attn_flash<<<dim3(S_/64, NH_, B_), 128, 0, stream>>>(qkv, att);

  // 4) x2 = (input + att @ Wo) * mask  -> f32
  gemm_wmma<2><<<dim3(4, 32), 256, 0, stream>>>(att, H_, wo_t, x2, H_,
      ROWS, H_, H_, input, mask, 1.0f);

  // 5) y = LN(x2) -> bf16
  ln_rows<<<ROWS, 256, 0, stream>>>(x2, mask, y, 0);

  // 6) h1 = gelu(y @ W1) -> bf16
  gemm_wmma<1><<<dim3(16, 32), 256, 0, stream>>>(y, H_, w1_t, h1, MLP_,
      ROWS, MLP_, H_, nullptr, nullptr, 1.0f);

  // 7) out = h1 @ W2 + x2 -> f32
  gemm_wmma<3><<<dim3(4, 32), 256, 0, stream>>>(h1, MLP_, w2_t, d_out, H_,
      ROWS, H_, MLP_, x2, nullptr, 1.0f);
}
